// MultiHeadSelfAttention2D_76785425318535
// MI455X (gfx1250) — matrix operand order repaired and checked by reading, not compiled, not run on hardware
//
#include <hip/hip_runtime.h>
#include <stdint.h>

typedef _Float16 f16;
typedef _Float16 v16h __attribute__((ext_vector_type(16)));
typedef _Float16 v8h  __attribute__((ext_vector_type(8)));
typedef float    v8f  __attribute__((ext_vector_type(8)));

union V16U { v16h v; v8h h[2]; };

#define A_  4
#define B_  4
#define C_  64
#define T_  1000
#define F_  65
#define E_  4
#define D_  16
#define EF  260      // E*F
#define EFP 288      // EF padded to 9*32 (WMMA K-chunks)
#define DF  1040     // D*F
#define TP  1008     // 63*16 row tiles
#define SP  1024     // s padded to 32*32 for P@V
#define NH  16       // A*B head-batch problems
#define NTT 63       // t tiles

// ---------------------------------------------------------------- zero fill
__global__ void zero_kernel(uint32_t* __restrict__ p, size_t n) {
  size_t i = (size_t)blockIdx.x * blockDim.x + threadIdx.x;
  size_t stride = (size_t)gridDim.x * blockDim.x;
  for (; i < n; i += stride) p[i] = 0u;
}

// -------------------------------------------- QKV proj + PReLU + LN + pack
__global__ __launch_bounds__(256) void proj_kernel(
    const float* __restrict__ x,
    const float* __restrict__ Wq, const float* __restrict__ bq, const float* __restrict__ aq,
    const float* __restrict__ gq, const float* __restrict__ hq,
    const float* __restrict__ Wk, const float* __restrict__ bk, const float* __restrict__ ak,
    const float* __restrict__ gk, const float* __restrict__ hk,
    const float* __restrict__ Wv, const float* __restrict__ bv, const float* __restrict__ av,
    const float* __restrict__ gv, const float* __restrict__ hv,
    f16* __restrict__ Qg, f16* __restrict__ Kg, f16* __restrict__ Vt) {
  __shared__ float xs[C_ * F_];
  __shared__ float qy[A_ * E_ * F_];
  __shared__ float ky[A_ * E_ * F_];
  __shared__ float vy[A_ * D_ * F_];
  __shared__ float gmu[12], ginv[12];
  const int tid = threadIdx.x;
  const int lane = tid & 31, wave = tid >> 5;
  const int t = blockIdx.x % T_, b = blockIdx.x / T_;

  for (int i = tid; i < C_ * F_; i += 256)
    xs[i] = x[(((size_t)b * C_ + i / F_) * T_ + t) * F_ + i % F_];
  __syncthreads();

  for (int i = tid; i < A_ * E_ * F_; i += 256) {
    int a = i / EF, r = i % EF, o = r / F_, f = r % F_;
    const float* wq = Wq + (a * E_ + o) * C_;
    const float* wk = Wk + (a * E_ + o) * C_;
    float sq = bq[a * E_ + o], sk = bk[a * E_ + o];
    for (int c = 0; c < C_; ++c) { float xv = xs[c * F_ + f]; sq += wq[c] * xv; sk += wk[c] * xv; }
    qy[i] = sq >= 0.f ? sq : aq[a] * sq;
    ky[i] = sk >= 0.f ? sk : ak[a] * sk;
  }
  for (int i = tid; i < A_ * D_ * F_; i += 256) {
    int a = i / DF, r = i % DF, d = r / F_, f = r % F_;
    const float* wv = Wv + (a * D_ + d) * C_;
    float s = bv[a * D_ + d];
    for (int c = 0; c < C_; ++c) s += wv[c] * xs[c * F_ + f];
    vy[i] = s >= 0.f ? s : av[a] * s;
  }
  __syncthreads();

  // 12 LN groups: 0-3 Q, 4-7 K, 8-11 V
  for (int g = wave; g < 12; g += 8) {
    const float* buf; int sz;
    if (g < 4)      { buf = qy + g * EF;        sz = EF; }
    else if (g < 8) { buf = ky + (g - 4) * EF;  sz = EF; }
    else            { buf = vy + (g - 8) * DF;  sz = DF; }
    float s = 0.f, s2 = 0.f;
    for (int i = lane; i < sz; i += 32) { float v = buf[i]; s += v; s2 += v * v; }
    for (int off = 16; off > 0; off >>= 1) {
      s += __shfl_xor(s, off, 32); s2 += __shfl_xor(s2, off, 32);
    }
    if (lane == 0) {
      float mu = s / sz, var = s2 / sz - mu * mu;
      gmu[g] = mu; ginv[g] = rsqrtf(var + 1e-5f);
    }
  }
  __syncthreads();

  for (int i = tid; i < A_ * E_ * F_; i += 256) {
    int a = i / EF, ef = i % EF;
    int nh = a * B_ + b;
    float q = (qy[i] - gmu[a]) * ginv[a] * gq[i] + hq[i];
    float k = (ky[i] - gmu[4 + a]) * ginv[4 + a] * gk[i] + hk[i];
    Qg[((size_t)nh * TP + t) * EFP + ef] = (f16)q;
    Kg[((size_t)nh * TP + t) * EFP + ef] = (f16)k;
  }
  for (int i = tid; i < A_ * D_ * F_; i += 256) {
    int a = i / DF, df = i % DF;
    int nh = a * B_ + b;
    float v = (vy[i] - gmu[8 + a]) * ginv[8 + a] * gv[i] + hv[i];
    Vt[((size_t)nh * DF + df) * SP + t] = (f16)v;   // transposed [df][s]
  }
}

// ---------------------------------------------------- attention with WMMA
__global__ __launch_bounds__(512) void attn_kernel(
    const f16* __restrict__ Qg, const f16* __restrict__ Kg,
    const f16* __restrict__ Vt, float* __restrict__ attnout) {
  extern __shared__ char smem[];
  f16*   Qs   = (f16*)smem;                                    // 16*288 f16
  float* Ss   = (float*)(smem + 16 * EFP * 2);                 // 16*1024 f32
  f16*   Ps   = (f16*)(smem + 16 * EFP * 2 + 16 * SP * 4);     // 16*1024 f16
  float* rinv = (float*)(smem + 16 * EFP * 2 + 16 * SP * 4 + 16 * SP * 2);

  const int tid  = threadIdx.x;
  const int lane = tid & 31;
  const int wave = tid >> 5;            // 16 waves
  const int col  = lane & 15;           // C/D column, A/B row/col within tile
  const int half = lane >> 4;
  const int ttile = blockIdx.x;         // 0..62
  const int n     = blockIdx.y;         // 0..15 head-batch
  const int tbase = ttile * 16;

  // load Q row-tile to LDS
  for (int i = tid; i < 16 * EFP; i += 512) {
    int r = i / EFP, ef = i % EFP;
    Qs[i] = Qg[((size_t)n * TP + (tbase + r)) * EFP + ef];
  }
  __syncthreads();

  const float scale = 0.06201736729f;   // 1/sqrt(E*F)=1/sqrt(260)

  // ---- Phase 1: S = Q @ K^T (scaled, masked)
  for (int st = wave; st < NTT; st += 16) {
    v8f acc = {};
    const int srow = st * 16 + col;     // this lane's B-column = K row
    const f16* kp = Kg + ((size_t)n * TP + srow) * EFP + half * 8;
#pragma unroll
    for (int kc = 0; kc < EFP / 32; ++kc) {
      int efb = kc * 32;
      V16U af, bf;
      af.h[0] = *(const v8h*)(Qs + col * EFP + efb + half * 8);        // A layout
      af.h[1] = *(const v8h*)(Qs + col * EFP + efb + 16 + half * 8);
      bf.h[0] = *(const v8h*)(kp + efb);                               // B layout
      bf.h[1] = *(const v8h*)(kp + efb + 16);
      acc = __builtin_amdgcn_wmma_f32_16x16x32_f16(false, af.v, false, bf.v,
                                                   (short)0, acc, false, false);
    }
#pragma unroll
    for (int r = 0; r < 8; ++r) {
      int m = r + 8 * half;             // D row
      int sg = st * 16 + col;           // D col = global s
      Ss[m * SP + sg] = (sg < T_) ? acc[r] * scale : -1e30f;
    }
  }
  __syncthreads();

  // ---- Phase 2: softmax, one wave per row
  {
    int m = wave;
    float mx = -3.0e38f;
    for (int c2 = lane; c2 < TP; c2 += 32) mx = fmaxf(mx, Ss[m * SP + c2]);
#pragma unroll
    for (int off = 16; off > 0; off >>= 1) mx = fmaxf(mx, __shfl_xor(mx, off, 32));
    float sum = 0.f;
    for (int c2 = lane; c2 < SP; c2 += 32) {
      float p = 0.f;
      if (c2 < TP) p = __expf(Ss[m * SP + c2] - mx);
      Ps[m * SP + c2] = (f16)p;
      sum += p;
    }
#pragma unroll
    for (int off = 16; off > 0; off >>= 1) sum += __shfl_xor(sum, off, 32);
    if (lane == 0) rinv[m] = 1.0f / sum;
  }
  __syncthreads();

  // ---- Phase 3: out = P @ V, scatter into [B,C,T,F]
  const int a_head = n >> 2;            // n = a*B + b (B=4)
  const int b_idx  = n & 3;
  for (int nt = wave; nt < (DF + 15) / 16; nt += 16) {   // 65 df tiles
    int dfb = nt * 16;
    const int df = dfb + col;           // < 1040 always
    const f16* vp = Vt + ((size_t)n * DF + df) * SP + half * 8;
    v8f acc = {};
#pragma unroll 4
    for (int kc = 0; kc < SP / 32; ++kc) {
      int sb = kc * 32;
      V16U af, bf;
      af.h[0] = *(const v8h*)(Ps + col * SP + sb + half * 8);
      af.h[1] = *(const v8h*)(Ps + col * SP + sb + 16 + half * 8);
      bf.h[0] = *(const v8h*)(vp + sb);
      bf.h[1] = *(const v8h*)(vp + sb + 16);
      acc = __builtin_amdgcn_wmma_f32_16x16x32_f16(false, af.v, false, bf.v,
                                                   (short)0, acc, false, false);
    }
    int d = df / F_, f = df % F_;
    int cch = a_head * D_ + d;
#pragma unroll
    for (int r = 0; r < 8; ++r) {
      int m = r + 8 * half;
      int t = tbase + m;
      if (t < T_)
        attnout[(((size_t)b_idx * C_ + cch) * T_ + t) * F_ + f] = acc[r] * rinv[m];
    }
  }
}

// ----------------------------------- out proj + PReLU + LN + residual
__global__ __launch_bounds__(256) void outproj_kernel(
    const float* __restrict__ attnout, const float* __restrict__ x,
    const float* __restrict__ Wo, const float* __restrict__ bo,
    const float* __restrict__ ao, const float* __restrict__ go,
    const float* __restrict__ ho, float* __restrict__ out) {
  __shared__ float as2[C_ * F_];
  __shared__ float ys[C_ * F_];
  __shared__ float wsum[8], wsq[8];
  __shared__ float smu, sinv;
  const int tid = threadIdx.x, lane = tid & 31, wave = tid >> 5;
  const int t = blockIdx.x % T_, b = blockIdx.x / T_;

  for (int i = tid; i < C_ * F_; i += 256)
    as2[i] = attnout[(((size_t)b * C_ + i / F_) * T_ + t) * F_ + i % F_];
  __syncthreads();
  const float aos = ao[0];
  for (int i = tid; i < C_ * F_; i += 256) {
    int o = i / F_, f = i % F_;
    float s = bo[o];
    const float* w = Wo + o * C_;
    for (int c = 0; c < C_; ++c) s += w[c] * as2[c * F_ + f];
    ys[i] = s >= 0.f ? s : aos * s;
  }
  __syncthreads();
  float s = 0.f, s2 = 0.f;
  for (int i = tid; i < C_ * F_; i += 256) { float v = ys[i]; s += v; s2 += v * v; }
  for (int off = 16; off > 0; off >>= 1) {
    s += __shfl_xor(s, off, 32); s2 += __shfl_xor(s2, off, 32);
  }
  if (lane == 0) { wsum[wave] = s; wsq[wave] = s2; }
  __syncthreads();
  if (tid == 0) {
    float ts = 0.f, tq = 0.f;
    for (int w2 = 0; w2 < 8; ++w2) { ts += wsum[w2]; tq += wsq[w2]; }
    float mu = ts / (C_ * F_), var = tq / (C_ * F_) - mu * mu;
    smu = mu; sinv = rsqrtf(var + 1e-5f);
  }
  __syncthreads();
  for (int i = tid; i < C_ * F_; i += 256) {
    int o = i / F_, f = i % F_;
    size_t idx = (((size_t)b * C_ + o) * T_ + t) * F_ + f;
    out[idx] = (ys[i] - smu) * sinv * go[i] + ho[i] + x[idx];
  }
}

// ------------------------------------------------------------------ launch
extern "C" void kernel_launch(void* const* d_in, const int* in_sizes, int n_in,
                              void* d_out, int out_size, void* d_ws, size_t ws_size,
                              hipStream_t stream) {
  (void)in_sizes; (void)n_in; (void)out_size; (void)ws_size;
  const float* x  = (const float*)d_in[0];
  const float* Wq = (const float*)d_in[1];
  const float* bq = (const float*)d_in[2];
  const float* aq = (const float*)d_in[3];
  const float* gq = (const float*)d_in[4];
  const float* hq = (const float*)d_in[5];
  const float* Wk = (const float*)d_in[6];
  const float* bk = (const float*)d_in[7];
  const float* ak = (const float*)d_in[8];
  const float* gk = (const float*)d_in[9];
  const float* hk = (const float*)d_in[10];
  const float* Wv = (const float*)d_in[11];
  const float* bv = (const float*)d_in[12];
  const float* av = (const float*)d_in[13];
  const float* gv = (const float*)d_in[14];
  const float* hv = (const float*)d_in[15];
  const float* Wo = (const float*)d_in[16];
  const float* bo = (const float*)d_in[17];
  const float* ao = (const float*)d_in[18];
  const float* go = (const float*)d_in[19];
  const float* ho = (const float*)d_in[20];

  const size_t QB = (size_t)NH * TP * EFP * sizeof(f16);   // 9,289,728 B
  const size_t VB = (size_t)NH * DF * SP * sizeof(f16);    // 34,078,720 B
  char* ws = (char*)d_ws;
  f16* Qg = (f16*)ws;
  f16* Kg = (f16*)(ws + QB);
  f16* Vt = (f16*)(ws + 2 * QB);
  float* attnout = (float*)(ws + 2 * QB + VB);

  zero_kernel<<<2048, 256, 0, stream>>>((uint32_t*)ws, (2 * QB + VB) / 4);
  proj_kernel<<<B_ * T_, 256, 0, stream>>>(x, Wq, bq, aq, gq, hq, Wk, bk, ak, gk, hk,
                                           Wv, bv, av, gv, hv, Qg, Kg, Vt);
  const int SMEM = 16 * EFP * 2 + 16 * SP * 4 + 16 * SP * 2 + 64;  // 107,584 B
  hipFuncSetAttribute((const void*)attn_kernel,
                      hipFuncAttributeMaxDynamicSharedMemorySize, SMEM);
  attn_kernel<<<dim3(NTT, NH), 512, SMEM, stream>>>(Qg, Kg, Vt, attnout);
  outproj_kernel<<<B_ * T_, 256, 0, stream>>>(attnout, x, Wo, bo, ao, go, ho, (float*)d_out);
}
